// JambaMoE_70248485094063
// MI455X (gfx1250) — matrix operand order repaired and checked by reading, not compiled, not run on hardware
//
#include <hip/hip_runtime.h>
#include <math.h>

// ---------------- problem constants ----------------
#define T_TOK 2048
#define HID   1024
#define FFNSZ 2816
#define NEXP  8
#define CAP   2048   // max assignments per expert (top-k indices distinct -> <= T)
#define MT    32     // tokens per expert tile (2 WMMA M-tiles)
#define FCH   1408   // FFN chunk (FFNSZ/2) kept in LDS at a time

// ---------------- vector types ----------------
typedef __attribute__((ext_vector_type(16))) __bf16 v16bf;
typedef __attribute__((ext_vector_type(8)))  __bf16 v8bf;
typedef __attribute__((ext_vector_type(4)))  __bf16 v4bf;
typedef __attribute__((ext_vector_type(8)))  float  v8f;

static __device__ inline v16bf frag_cat(v8bf lo, v8bf hi) {
  return __builtin_shufflevector(lo, hi, 0,1,2,3,4,5,6,7,8,9,10,11,12,13,14,15);
}

// ---------------- init: zero output + expert counters ----------------
__global__ void init_kernel(float* __restrict__ out, size_t n, int* __restrict__ counts) {
  size_t idx = (size_t)blockIdx.x * blockDim.x + threadIdx.x;
  size_t stride = (size_t)gridDim.x * blockDim.x;
  for (size_t i = idx; i < n; i += stride) out[i] = 0.0f;
  if (idx < NEXP) counts[idx] = 0;
}

// ---------------- fp32 -> bf16 conversion (vectorized) ----------------
__global__ void cvt_kernel(const float4* __restrict__ src, v4bf* __restrict__ dst, size_t n4) {
  size_t idx = (size_t)blockIdx.x * blockDim.x + threadIdx.x;
  size_t stride = (size_t)gridDim.x * blockDim.x;
  for (size_t i = idx; i < n4; i += stride) {
    float4 v = src[i];
    v4bf o;
    o[0] = (__bf16)v.x; o[1] = (__bf16)v.y; o[2] = (__bf16)v.z; o[3] = (__bf16)v.w;
    dst[i] = o;
  }
}

// ---------------- router: logits -> top-2 -> softmax -> expert lists ----------------
__global__ __launch_bounds__(256) void router_kernel(const float* __restrict__ x,
                                                     const float* __restrict__ gw,
                                                     int* __restrict__ counts,
                                                     int* __restrict__ lists,
                                                     float* __restrict__ wts) {
  const int t    = blockIdx.x;
  const int lane = threadIdx.x & 31;
  const int wid  = threadIdx.x >> 5;    // 8 waves == 8 experts
  __shared__ float lg[NEXP];

  float p = 0.0f;
  const float* xr = x  + (size_t)t   * HID;
  const float* gr = gw + (size_t)wid * HID;
  for (int i = lane; i < HID; i += 32) p += xr[i] * gr[i];
  for (int off = 16; off > 0; off >>= 1) p += __shfl_xor(p, off, 32);
  if (lane == 0) lg[wid] = p;
  __syncthreads();

  if (threadIdx.x == 0) {
    // top-2 (first occurrence on ties, matching jax.lax.top_k)
    int i1 = 0; float v1 = lg[0];
    for (int e2 = 1; e2 < NEXP; ++e2) if (lg[e2] > v1) { v1 = lg[e2]; i1 = e2; }
    int i2 = -1; float v2 = -3.0e38f;
    for (int e2 = 0; e2 < NEXP; ++e2) { if (e2 == i1) continue; if (lg[e2] > v2) { v2 = lg[e2]; i2 = e2; } }
    float b  = __expf(v2 - v1);           // softmax over [v1, v2], v1 is max
    float wA = 1.0f / (1.0f + b);
    float wB = b * wA;
    int s1 = atomicAdd(&counts[i1], 1);
    lists[i1 * CAP + s1] = t;  wts[i1 * CAP + s1] = wA;
    int s2 = atomicAdd(&counts[i2], 1);
    lists[i2 * CAP + s2] = t;  wts[i2 * CAP + s2] = wB;
  }
}

// ---------------- fused expert FFN over gathered token tiles ----------------
// Block: 256 threads (8 waves), tile of MT=32 gathered tokens, one expert.
// FFN processed in two LDS-resident chunks; per chunk:
//   phase 1: act = silu(x*w1^T) * (x*w3^T)   (bf16 WMMA, 2 M-tiles share each B)
//   phase 2: y += act_chunk * w2_chunk^T, scaled + atomically accumulated.
__global__ __launch_bounds__(256) void moe_kernel(const __bf16* __restrict__ xb,
                                                  const __bf16* __restrict__ w1b,
                                                  const __bf16* __restrict__ w3b,
                                                  const __bf16* __restrict__ w2b,
                                                  const int*   __restrict__ counts,
                                                  const int*   __restrict__ lists,
                                                  const float* __restrict__ wts,
                                                  float* __restrict__ out) {
  const int e    = blockIdx.y;
  const int tile = blockIdx.x;
  const int cnt  = counts[e];
  if (tile * MT >= cnt) return;           // block-uniform early exit

  __shared__ __bf16 xs  [MT][HID];        // gathered x tile (bf16)     64 KB
  __shared__ __bf16 actc[MT][FCH];        // silu(g)*u chunk (bf16)     88 KB
  __shared__ int    tok[MT];
  __shared__ float  twt[MT];

  const int tid  = threadIdx.x;
  const int lane = tid & 31;
  const int wid  = tid >> 5;
  const int n    = lane & 15;             // N index (B/C/D), M index (A)
  const int half = lane >> 4;             // selects K-run / M-half
  const int m    = n;

  if (tid < MT) {
    int slot = tile * MT + tid;
    if (slot < cnt) { tok[tid] = lists[e * CAP + slot]; twt[tid] = wts[e * CAP + slot]; }
    else            { tok[tid] = 0;                     twt[tid] = 0.0f; }
  }
  __syncthreads();

  // gather MT token rows of bf16 x into LDS (16B chunks)
  for (int i = tid; i < MT * (HID / 8); i += 256) {
    int row = i >> 7, c8 = i & 127;
    *(v8bf*)&xs[row][c8 * 8] = *(const v8bf*)&xb[(size_t)tok[row] * HID + c8 * 8];
  }
  __syncthreads();

  const __bf16* w1e = w1b + (size_t)e * FFNSZ * HID;
  const __bf16* w3e = w3b + (size_t)e * FFNSZ * HID;
  const __bf16* w2e = w2b + (size_t)e * HID * FFNSZ;

  for (int c = 0; c < FFNSZ / FCH; ++c) {
    const int fbase = c * FCH;

    // ---------- phase 1: gate/up projections + SiLU for this chunk ----------
    for (int ft = wid; ft < FCH / 16; ft += 8) {   // 88 f-tiles / 8 waves = 11
      const int f0 = fbase + ft * 16;
      v8f cg0 = {}, cu0 = {}, cg1 = {}, cu1 = {};
      const __bf16* w1row = w1e + (size_t)(f0 + n) * HID;
      const __bf16* w3row = w3e + (size_t)(f0 + n) * HID;
      for (int h0 = 0; h0 < HID; h0 += 32) {
        // A fragments: lane holds K = [h0+8*half,+8) and [h0+16+8*half,+8)
        v16bf A0 = frag_cat(*(const v8bf*)&xs[m][h0 + 8 * half],
                            *(const v8bf*)&xs[m][h0 + 16 + 8 * half]);
        v16bf A1 = frag_cat(*(const v8bf*)&xs[16 + m][h0 + 8 * half],
                            *(const v8bf*)&xs[16 + m][h0 + 16 + 8 * half]);
        // B fragments: lane holds K = [h0+8*half,+8) and [h0+16+8*half,+8) of row f0+n
        v16bf B1 = frag_cat(*(const v8bf*)(w1row + h0 + 8 * half),
                            *(const v8bf*)(w1row + h0 + 16 + 8 * half));
        cg0 = __builtin_amdgcn_wmma_f32_16x16x32_bf16(false, A0, false, B1, (short)0, cg0, false, false);
        cg1 = __builtin_amdgcn_wmma_f32_16x16x32_bf16(false, A1, false, B1, (short)0, cg1, false, false);
        v16bf B3 = frag_cat(*(const v8bf*)(w3row + h0 + 8 * half),
                            *(const v8bf*)(w3row + h0 + 16 + 8 * half));
        cu0 = __builtin_amdgcn_wmma_f32_16x16x32_bf16(false, A0, false, B3, (short)0, cu0, false, false);
        cu1 = __builtin_amdgcn_wmma_f32_16x16x32_bf16(false, A1, false, B3, (short)0, cu1, false, false);
      }
      #pragma unroll
      for (int r = 0; r < 8; ++r) {
        float g0 = cg0[r], u0 = cu0[r];
        float g1 = cg1[r], u1 = cu1[r];
        float s0 = g0 / (1.0f + __expf(-g0));     // SiLU
        float s1 = g1 / (1.0f + __expf(-g1));
        actc[r + 8 * half     ][ft * 16 + n] = (__bf16)(s0 * u0);
        actc[r + 8 * half + 16][ft * 16 + n] = (__bf16)(s1 * u1);
      }
    }
    __syncthreads();

    // ---------- phase 2: partial down projection + scatter-accumulate ----------
    for (int ht = wid; ht < HID / 16; ht += 8) {   // 64 h-tiles / 8 waves = 8
      const int h0 = ht * 16;
      v8f cy0 = {}, cy1 = {};
      const __bf16* w2row = w2e + (size_t)(h0 + n) * FFNSZ + fbase;
      for (int k0 = 0; k0 < FCH; k0 += 32) {
        v16bf A0 = frag_cat(*(const v8bf*)&actc[m][k0 + 8 * half],
                            *(const v8bf*)&actc[m][k0 + 16 + 8 * half]);
        v16bf A1 = frag_cat(*(const v8bf*)&actc[16 + m][k0 + 8 * half],
                            *(const v8bf*)&actc[16 + m][k0 + 16 + 8 * half]);
        v16bf B  = frag_cat(*(const v8bf*)(w2row + k0 + 8 * half),
                            *(const v8bf*)(w2row + k0 + 16 + 8 * half));
        cy0 = __builtin_amdgcn_wmma_f32_16x16x32_bf16(false, A0, false, B, (short)0, cy0, false, false);
        cy1 = __builtin_amdgcn_wmma_f32_16x16x32_bf16(false, A1, false, B, (short)0, cy1, false, false);
      }
      #pragma unroll
      for (int r = 0; r < 8; ++r) {
        int m0 = r + 8 * half;
        int m1 = m0 + 16;
        atomicAdd(&out[(size_t)tok[m0] * HID + h0 + n], cy0[r] * twt[m0]);
        atomicAdd(&out[(size_t)tok[m1] * HID + h0 + n], cy1[r] * twt[m1]);
      }
    }
    __syncthreads();   // actc reused by next chunk
  }
}

// ---------------- host launcher ----------------
extern "C" void kernel_launch(void* const* d_in, const int* in_sizes, int n_in,
                              void* d_out, int out_size, void* d_ws, size_t ws_size,
                              hipStream_t stream) {
  (void)in_sizes; (void)n_in; (void)out_size; (void)ws_size;
  const float* x  = (const float*)d_in[0];
  const float* gw = (const float*)d_in[1];
  const float* w1 = (const float*)d_in[2];
  const float* w3 = (const float*)d_in[3];
  const float* w2 = (const float*)d_in[4];
  float* out = (float*)d_out;

  const size_t XEL = (size_t)T_TOK * HID;          // x elements
  const size_t WEL = (size_t)NEXP * FFNSZ * HID;   // per-weight elements

  char* ws = (char*)d_ws;
  size_t off = 0;
  __bf16* xb  = (__bf16*)(ws + off); off += XEL * 2;
  __bf16* w1b = (__bf16*)(ws + off); off += WEL * 2;
  __bf16* w3b = (__bf16*)(ws + off); off += WEL * 2;
  __bf16* w2b = (__bf16*)(ws + off); off += WEL * 2;
  int*    counts = (int*)(ws + off); off += 256;   // 8 ints, padded
  int*    lists  = (int*)(ws + off); off += (size_t)NEXP * CAP * sizeof(int);
  float*  wtsp   = (float*)(ws + off); off += (size_t)NEXP * CAP * sizeof(float);

  init_kernel<<<1024, 256, 0, stream>>>(out, XEL, counts);
  cvt_kernel<<<2048, 256, 0, stream>>>((const float4*)x,  (v4bf*)xb,  XEL / 4);
  cvt_kernel<<<8192, 256, 0, stream>>>((const float4*)w1, (v4bf*)w1b, WEL / 4);
  cvt_kernel<<<8192, 256, 0, stream>>>((const float4*)w3, (v4bf*)w3b, WEL / 4);
  cvt_kernel<<<8192, 256, 0, stream>>>((const float4*)w2, (v4bf*)w2b, WEL / 4);
  router_kernel<<<T_TOK, 256, 0, stream>>>(x, gw, counts, lists, wtsp);
  moe_kernel<<<dim3(CAP / MT, NEXP), 256, 0, stream>>>(xb, w1b, w3b, w2b,
                                                       counts, lists, wtsp, out);
}
